// TemporalSelfAttention_update_2860448219660
// MI455X (gfx1250) — compile-verified, matrix operand order checked by reading, not run on hardware
//
#include <hip/hip_runtime.h>
#include <hip/hip_bf16.h>

typedef __attribute__((ext_vector_type(16))) _Float16 v16h;
typedef __attribute__((ext_vector_type(4)))  _Float16 v4h;
typedef __attribute__((ext_vector_type(8)))  float    v8f;

#define NQN   40000
#define DIM   256
#define NHD   8
#define DHD   32
#define HH    200
#define WW    200

// ---------------------------------------------------------------------------
// Prep: qwh f16 = [query | query+query_pos]  (row stride 512)
// ---------------------------------------------------------------------------
__global__ void prep_qwh(const float* __restrict__ query,
                         const float* __restrict__ qpos,
                         _Float16* __restrict__ qwh) {
  int idx = blockIdx.x * blockDim.x + threadIdx.x;
  if (idx >= NQN * DIM) return;
  int q = idx >> 8, c = idx & 255;
  float qu = query[idx];
  qwh[(size_t)q * 512 + c]       = (_Float16)qu;            // history_bev == query
  qwh[(size_t)q * 512 + 256 + c] = (_Float16)(qu + qpos[idx]);
}

// dst (f16, N x K row-major) = transpose of src (f32, K x N row-major)
__global__ void transpose_f16(const float* __restrict__ src,
                              _Float16* __restrict__ dst, int K, int N) {
  int idx = blockIdx.x * blockDim.x + threadIdx.x;
  if (idx >= N * K) return;
  int n = idx / K, k = idx - n * K;
  dst[idx] = (_Float16)src[(size_t)k * N + n];
}

__global__ void concat_bias(const float* __restrict__ a, const float* __restrict__ b,
                            float* __restrict__ dst, int na, int nb) {
  int i = blockIdx.x * blockDim.x + threadIdx.x;
  if (i < na) dst[i] = a[i];
  else if (i < na + nb) dst[i] = b[i - na];
}

// ---------------------------------------------------------------------------
// WMMA GEMM: C(M x N) = A(M x K, f16, lda) * Bt(N x K, f16)^T + bias(N)
// One wave per 16 x 64 output tile. K is a compile-time constant so the
// k-loop fully unrolls into straight-line SSA: the scheduler hoists the
// immediate-offset b128 load clauses across WMMA groups with no register
// rotation moves.
// ---------------------------------------------------------------------------
template <int K>
__global__ __launch_bounds__(32)
void gemm_wmma_f16(const _Float16* __restrict__ A, int lda,
                   const float* __restrict__ bias,
                   const _Float16* __restrict__ Bt,   // N x K row-major
                   float* __restrict__ C, int ldc) {
  const int lane = threadIdx.x & 31;
  const int m0   = blockIdx.x * 16;
  const int n0   = blockIdx.y * 64;
  const int r15  = lane & 15;
  const int kb   = (lane >> 4) * 16;

  v8f acc[4] = {};
  const _Float16* Arow = A + (size_t)(m0 + r15) * lda + kb;
  const _Float16* Brow[4];
#pragma unroll
  for (int j = 0; j < 4; ++j)
    Brow[j] = Bt + (size_t)(n0 + j * 16 + r15) * K + kb;

#pragma unroll
  for (int k = 0; k < K; k += 32) {
    v16h a = *(const v16h*)(Arow + k);
    v16h b0 = *(const v16h*)(Brow[0] + k);
    v16h b1 = *(const v16h*)(Brow[1] + k);
    v16h b2 = *(const v16h*)(Brow[2] + k);
    v16h b3 = *(const v16h*)(Brow[3] + k);
    acc[0] = __builtin_amdgcn_wmma_f32_16x16x32_f16(false, a, false, b0, (short)0, acc[0], false, false);
    acc[1] = __builtin_amdgcn_wmma_f32_16x16x32_f16(false, a, false, b1, (short)0, acc[1], false, false);
    acc[2] = __builtin_amdgcn_wmma_f32_16x16x32_f16(false, a, false, b2, (short)0, acc[2], false, false);
    acc[3] = __builtin_amdgcn_wmma_f32_16x16x32_f16(false, a, false, b3, (short)0, acc[3], false, false);
  }

  const int outrow = m0 + (lane >> 4) * 8;
#pragma unroll
  for (int j = 0; j < 4; ++j) {
    const int n = n0 + j * 16 + r15;
    const float bv = bias[n];
#pragma unroll
    for (int r = 0; r < 8; ++r)
      C[(size_t)(outrow + r) * ldc + n] = acc[j][r] + bv;
  }
}

// ---------------------------------------------------------------------------
// Stage 1: lane-parallel weight/address computation.
// One thread per (q, h, queue, p) task: task c = h*8 + queue*4 + p (64 per q).
// aw logit column = 128 + c ; so columns = 2c (x), 2c+1 (y).
// Softmax over the 4 points = aligned 4-lane group -> two __shfl_xor rounds.
// Emits: abuf = clampedBase(16b) | dx<<16 | dy<<17 ; wbuf = 4 f16 corner
// weights (validity and 0.5 queue-mean folded in). Border clipping matches
// the reference: OOB corners get weight 0; the clipped neighbor collapses to
// dx/dy = 0 so the valid corner still reads the correct texel.
// ---------------------------------------------------------------------------
__global__ __launch_bounds__(256)
void deform_weights(const float* __restrict__ soaw,  // (NQ, 192)
                    const float* __restrict__ rp,    // (NQ, 2)
                    unsigned int* __restrict__ abuf, // (NQ*64)
                    _Float16* __restrict__ wbuf) {   // (NQ*64, 4)
  const int tid = blockIdx.x * 256 + threadIdx.x;    // < NQN*64
  const int q = tid >> 6;
  const int c = tid & 63;
  const float* row = soaw + (size_t)q * 192;

  float l = row[128 + c];
  float m = l;
  m = fmaxf(m, __shfl_xor(m, 1));
  m = fmaxf(m, __shfl_xor(m, 2));
  float e = __expf(l - m);
  float s = e;
  s += __shfl_xor(s, 1);
  s += __shfl_xor(s, 2);
  float w = 0.5f * e / s;                            // 0.5 = queue mean

  float x = rp[q * 2 + 0] * (float)WW + row[2 * c]     - 0.5f;
  float y = rp[q * 2 + 1] * (float)HH + row[2 * c + 1] - 0.5f;
  float xf = floorf(x), yf = floorf(y);
  int   x0 = (int)xf,   y0 = (int)yf;
  float wx1 = x - xf, wx0 = 1.f - wx1;
  float wy1 = y - yf, wy0 = 1.f - wy1;

  float fx0 = (x0 >= 0     && x0 < WW)     ? wx0 : 0.f;
  float fx1 = (x0 + 1 >= 0 && x0 + 1 < WW) ? wx1 : 0.f;
  float fy0 = (y0 >= 0     && y0 < HH)     ? wy0 : 0.f;
  float fy1 = (y0 + 1 >= 0 && y0 + 1 < HH) ? wy1 : 0.f;

  int xb0 = min(max(x0, 0), WW - 1);
  int xb1 = min(max(x0 + 1, 0), WW - 1);
  int yb0 = min(max(y0, 0), HH - 1);
  int yb1 = min(max(y0 + 1, 0), HH - 1);

  abuf[tid] = (unsigned int)(yb0 * WW + xb0)
            | ((unsigned int)(xb1 - xb0) << 16)
            | ((unsigned int)(yb1 - yb0) << 17);
  v4h wv = { (_Float16)(w * fx0 * fy0), (_Float16)(w * fx1 * fy0),
             (_Float16)(w * fx0 * fy1), (_Float16)(w * fx1 * fy1) };
  *(v4h*)(wbuf + (size_t)tid * 4) = wv;
}

// ---------------------------------------------------------------------------
// Stage 2: gather. One wave per (q, h); lane = channel d (DH == 32).
// Per task: 2 descriptor loads + 4 coalesced 128B gathers + 4 FMAs.
// ---------------------------------------------------------------------------
__global__ __launch_bounds__(256)
void deform_gather(const float* __restrict__ v,          // (Nv, 256)
                   const unsigned int* __restrict__ abuf,
                   const _Float16* __restrict__ wbuf,
                   _Float16* __restrict__ outmid) {      // (NQ, 256)
  const int q    = blockIdx.x;
  const int h    = threadIdx.x >> 5;
  const int lane = threadIdx.x & 31;
  const float* vhl = v + h * DHD + lane;
  const int tbase = q * 64 + h * 8;

  float acc = 0.f;
#pragma unroll
  for (int t = 0; t < 8; ++t) {
    unsigned int u = abuf[tbase + t];
    v4h wv = *(const v4h*)(wbuf + (size_t)(tbase + t) * 4);
    int base = (int)(u & 0xFFFFu);
    int dx   = (int)((u >> 16) & 1u);
    int dyo  = ((u >> 17) & 1u) ? WW : 0;
    const float* p = vhl + (size_t)base * DIM;
    float g00 = p[0];
    float g01 = p[(size_t)dx * DIM];
    float g10 = p[(size_t)dyo * DIM];
    float g11 = p[(size_t)(dyo + dx) * DIM];
    acc += (float)wv.x * g00 + (float)wv.y * g01
         + (float)wv.z * g10 + (float)wv.w * g11;
  }
  outmid[(size_t)q * DIM + h * DHD + lane] = (_Float16)acc;
}

// ---------------------------------------------------------------------------
extern "C" void kernel_launch(void* const* d_in, const int* in_sizes, int n_in,
                              void* d_out, int out_size, void* d_ws, size_t ws_size,
                              hipStream_t stream) {
  (void)in_sizes; (void)n_in; (void)out_size; (void)ws_size;
  const float* query = (const float*)d_in[0];
  const float* qpos  = (const float*)d_in[1];
  const float* rp    = (const float*)d_in[2];
  const float* W_so  = (const float*)d_in[3];
  const float* b_so  = (const float*)d_in[4];
  const float* W_aw  = (const float*)d_in[5];
  const float* b_aw  = (const float*)d_in[6];
  const float* W_v   = (const float*)d_in[7];
  const float* b_v   = (const float*)d_in[8];
  const float* W_o   = (const float*)d_in[9];
  const float* b_o   = (const float*)d_in[10];
  float* out = (float*)d_out;

  char* ws = (char*)d_ws;                       // all offsets 256B-aligned
  _Float16*     qwh     = (_Float16*)    (ws + 0);          // 40000x512 f16
  _Float16*     Wv_t    = (_Float16*)    (ws + 40960000);   // 256x256 f16
  _Float16*     Wsoaw_t = (_Float16*)    (ws + 41091072);   // 192x512 f16
  _Float16*     Wo_t    = (_Float16*)    (ws + 41287680);   // 256x256 f16
  float*        b_soaw  = (float*)       (ws + 41418752);   // 192 f32
  float*        vbuf    = (float*)       (ws + 41419520);   // 40000x256 f32
  float*        soaw    = (float*)       (ws + 82379520);   // 40000x192 f32
  _Float16*     outmid  = (_Float16*)    (ws + 113099520);  // 40000x256 f16
  unsigned int* abuf    = (unsigned int*)(ws + 133579520);  // 2.56M u32
  _Float16*     wbuf    = (_Float16*)    (ws + 143819520);  // 2.56M x4 f16

  // --- prep (tiny, bandwidth-trivial) ---
  prep_qwh<<<(NQN * DIM + 255) / 256, 256, 0, stream>>>(query, qpos, qwh);
  transpose_f16<<<(256 * 256 + 255) / 256, 256, 0, stream>>>(W_v, Wv_t, 256, 256);
  transpose_f16<<<(128 * 512 + 255) / 256, 256, 0, stream>>>(W_so, Wsoaw_t, 512, 128);
  transpose_f16<<<( 64 * 512 + 255) / 256, 256, 0, stream>>>(W_aw, Wsoaw_t + 128 * 512, 512, 64);
  transpose_f16<<<(256 * 256 + 255) / 256, 256, 0, stream>>>(W_o, Wo_t, 256, 256);
  concat_bias<<<1, 256, 0, stream>>>(b_so, b_aw, b_soaw, 128, 64);

  // v = query @ W_v + b_v  (A = qwh cols 0..255, lda=512), N=256 -> grid.y=4
  gemm_wmma_f16<256><<<dim3(NQN / 16, 4), 32, 0, stream>>>(qwh, 512, b_v, Wv_t, vbuf, 256);
  // soaw = qwh @ [W_so|W_aw] + [b_so|b_aw], K=512, N=192 -> grid.y=3
  gemm_wmma_f16<512><<<dim3(NQN / 16, 3), 32, 0, stream>>>(qwh, 512, b_soaw, Wsoaw_t, soaw, 192);

  // deformable attention: lane-parallel weights, then coalesced gather
  deform_weights<<<(NQN * 64) / 256, 256, 0, stream>>>(soaw, rp, abuf, wbuf);
  deform_gather<<<NQN, 256, 0, stream>>>(vbuf, abuf, wbuf, outmid);

  // out = outmid @ W_o + b_o
  gemm_wmma_f16<256><<<dim3(NQN / 16, 4), 32, 0, stream>>>(outmid, 256, b_o, Wo_t, out, 256);
}
